// RiskGNN_89919435309313
// MI455X (gfx1250) — matrix operand order repaired and checked by reading, not compiled, not run on hardware
//
#include <hip/hip_runtime.h>
#include <math.h>

#define D 128

typedef __attribute__((ext_vector_type(16))) __bf16 v16bf;
typedef __attribute__((ext_vector_type(8)))  float  v8f;

__device__ __forceinline__ v8f wmma_bf16(v16bf a, v16bf b, v8f c) {
  // 8 args: (neg_a, A, neg_b, B, c_mod, C, reuse_a, reuse_b)
  return __builtin_amdgcn_wmma_f32_16x16x32_bf16(false, a, false, b,
                                                 (short)0, c, false, false);
}

// order-preserving f32 -> u32 encoding for atomic max
__device__ __forceinline__ unsigned fenc(float x) {
  unsigned u = __float_as_uint(x);
  return (u & 0x80000000u) ? ~u : (u | 0x80000000u);
}
__device__ __forceinline__ float fdec(unsigned u) {
  return (u & 0x80000000u) ? __uint_as_float(u & 0x7fffffffu)
                           : __uint_as_float(~u);
}

// ---------------------------------------------------------------------------
// Packing kernels.
// A rows (f32 row-major [M x 128]) -> bf16 rows in WMMA A-fragment lane order:
//   row layout: 8 groups g = kt*2 + khalf, each 16 bf16 covering
//   k = kt*32 + khalf*8 + {0..7, 16..23}   (ISA §7.12.2, 16-bit A 16x32)
// ---------------------------------------------------------------------------
__global__ void pack_a_kernel(const float* __restrict__ A, __bf16* __restrict__ pA,
                              long long total /* M*128 */) {
  long long t = (long long)blockIdx.x * blockDim.x + threadIdx.x;
  if (t >= total) return;
  int j = (int)(t & 127);
  long long row = t >> 7;
  int g = j >> 4, i = j & 15;
  int kt = g >> 1, kh = (g & 1) * 8;
  int k = kt * 32 + kh + ((i & 8) ? (i + 8) : i);
  pA[t] = (__bf16)A[row * D + k];
}

// W (f32 row-major [128 x 128], k-major) -> bf16 fragment order
//   pW[((nt*4 + kt)*32 + lane)*16 + i] = W[kb*128 + n]
//   n = nt*16 + (lane&15); kb = kt*32 + (lane>>4)*8 + (i<8 ? i : i+8)   (16-bit B 32x16)
__global__ void pack_w_kernel(const float* __restrict__ W, __bf16* __restrict__ pW) {
  int t = blockIdx.x * blockDim.x + threadIdx.x;   // 0..16383
  if (t >= D * D) return;
  int i    = t & 15;
  int lane = (t >> 4) & 31;
  int kt   = (t >> 9) & 3;
  int nt   = t >> 11;
  int n  = nt * 16 + (lane & 15);
  int kb = kt * 32 + (lane >> 4) * 8 + ((i & 8) ? (i + 8) : i);
  pW[t] = (__bf16)W[kb * D + n];
}

// ---------------------------------------------------------------------------
// GEMM: C[M x 128] = A[M x 128] @ W[128 x 128] + bias   (bf16 WMMA, f32 acc)
// block = 256 (8 waves); wave w -> N-tile w; blockIdx.x -> M-tile.
// 16 packed A rows staged in LDS once per block; B fragments are 32B
// contiguous loads from the packed weight image (L2-resident).
// ---------------------------------------------------------------------------
__global__ __launch_bounds__(256) void gemm_bias_kernel(
    const __bf16* __restrict__ pA, const __bf16* __restrict__ pW,
    const float* __restrict__ bias, float* __restrict__ C, int M) {
  __shared__ __attribute__((aligned(32))) __bf16 sA[16 * D];
  int tid = threadIdx.x;
  int m0  = blockIdx.x * 16;
  {
    int r   = tid >> 4;          // 0..15 : row in tile
    int seg = tid & 15;          // 16B segment of the 256B packed row
    int row = m0 + r; if (row >= M) row = M - 1;
    ((uint4*)sA)[r * 16 + seg] =
        ((const uint4*)(pA + (size_t)row * D))[seg];
  }
  __syncthreads();

  int lane = tid & 31;
  int wv   = tid >> 5;
  int n0   = wv * 16;
  v8f acc = {};
#pragma unroll
  for (int kt = 0; kt < 4; ++kt) {
    v16bf a = *(const v16bf*)(sA + (lane & 15) * D + (kt * 2 + (lane >> 4)) * 16);
    v16bf b = *(const v16bf*)(pW + (((size_t)wv * 4 + kt) * 32 + lane) * 16);
    acc = wmma_bf16(a, b, acc);
  }
  int n = n0 + (lane & 15);
  float bs = bias[n];
  int mbase = m0 + ((lane >> 4) ? 8 : 0);
  float* cp = C + (size_t)mbase * D + n;
  if (m0 + 16 <= M) {              // fast path: full tile, no guards
#pragma unroll
    for (int v = 0; v < 8; ++v) cp[(size_t)v * D] = acc[v] + bs;
  } else {
#pragma unroll
    for (int v = 0; v < 8; ++v)
      if (mbase + v < M) cp[(size_t)v * D] = acc[v] + bs;
  }
}

// ---------------------------------------------------------------------------
// Grouped per-relation edge GEMM over relation-sorted, 16-aligned-padded
// positions: res[p] = xlin[src[perm[p]]] @ wrel[r].
// The 16 gathered rows are staged in LDS once and shared by all 8 waves.
// Validity of padded tail rows is a register compare (p < offp[r]+cnt[r]).
// ---------------------------------------------------------------------------
__global__ __launch_bounds__(256) void rel_gemm_kernel(
    const __bf16* __restrict__ pxlin, const __bf16* __restrict__ pwrel_l,
    const int* __restrict__ srcs, const int* __restrict__ perm,
    const int* __restrict__ offp, const int* __restrict__ cnt,
    int Rn, float* __restrict__ res) {
  int p0 = blockIdx.x * 16;
  if (p0 >= offp[Rn]) return;                    // uniform per block
  int r = 0;
  while (r + 1 < Rn && p0 >= offp[r + 1]) ++r;   // offsets are 16-aligned
  int vend = offp[r] + cnt[r];                   // first padded (invalid) pos
  const __bf16* pW = pwrel_l + (size_t)r * D * D;

  __shared__ __attribute__((aligned(32))) __bf16 sA[16 * D];
  int tid = threadIdx.x;
  {
    int rr  = tid >> 4;
    int seg = tid & 15;
    int p   = p0 + rr;
    int row = (p < vend) ? srcs[perm[p]] : 0;    // padded slots: clamp
    ((uint4*)sA)[rr * 16 + seg] =
        ((const uint4*)(pxlin + (size_t)row * D))[seg];
  }
  __syncthreads();

  int lane = tid & 31;
  int wv   = tid >> 5;
  v8f acc = {};
#pragma unroll
  for (int kt = 0; kt < 4; ++kt) {
    v16bf a = *(const v16bf*)(sA + (lane & 15) * D + (kt * 2 + (lane >> 4)) * 16);
    v16bf b = *(const v16bf*)(pW + (((size_t)wv * 4 + kt) * 32 + lane) * 16);
    acc = wmma_bf16(a, b, acc);
  }
  int n = wv * 16 + (lane & 15);
  int mbase = (lane >> 4) ? 8 : 0;
  float* rp = res + (size_t)(p0 + mbase) * D + n;
  if (p0 + 16 <= vend) {           // fast path: whole tile valid
#pragma unroll
    for (int v = 0; v < 8; ++v) rp[(size_t)v * D] = acc[v];
  } else {
#pragma unroll
    for (int v = 0; v < 8; ++v)
      if (p0 + mbase + v < vend) rp[(size_t)v * D] = acc[v];
  }
}

// ---------------------------------------------------------------------------
// small utility / elementwise kernels
// ---------------------------------------------------------------------------
__global__ void fill_f32_kernel(float* p, long long n, float v) {
  long long i = (long long)blockIdx.x * blockDim.x + threadIdx.x;
  if (i < n) p[i] = v;
}
__global__ void fill_u32_kernel(unsigned* p, long long n, unsigned v) {
  long long i = (long long)blockIdx.x * blockDim.x + threadIdx.x;
  if (i < n) p[i] = v;
}

__global__ void count_kernel(const int* __restrict__ etype, int* cnt, int E) {
  long long e = (long long)blockIdx.x * blockDim.x + threadIdx.x;
  if (e < E) atomicAdd(&cnt[etype[e]], 1);
}
__global__ void offsets_kernel(const int* cnt, int* offp, int Rn) {
  if (threadIdx.x == 0 && blockIdx.x == 0) {
    int acc = 0; offp[0] = 0;
    for (int r = 0; r < Rn; ++r) { acc += (cnt[r] + 15) & ~15; offp[r + 1] = acc; }
  }
}
__global__ void scatter_kernel(const int* __restrict__ etype, int* cursor,
                               const int* __restrict__ offp, int* perm, int E) {
  long long e = (long long)blockIdx.x * blockDim.x + threadIdx.x;
  if (e >= E) return;
  int r = etype[e];
  int pos = atomicAdd(&cursor[r], 1);
  perm[offp[r] + pos] = (int)e;
}

__global__ void deg_kernel(const int* __restrict__ src, float* deg, int E) {
  long long e = (long long)blockIdx.x * blockDim.x + threadIdx.x;
  if (e < E) atomicAdd(&deg[src[e]], 1.0f);
}
__global__ void dinv_kernel(const float* deg, float* dinv, int N) {
  long long i = (long long)blockIdx.x * blockDim.x + threadIdx.x;
  if (i < N) dinv[i] = deg[i] > 0.f ? rsqrtf(fmaxf(deg[i], 1.f)) : 0.f;
}

// BN column stats: per-thread channel, 2 threads/channel, L2 atomics
__global__ __launch_bounds__(256) void bn_stats_kernel(
    const float* __restrict__ H, float* stat, int M) {
  int c = threadIdx.x & (D - 1);
  int half = threadIdx.x >> 7;
  float s = 0.f, s2 = 0.f;
  for (long long r = blockIdx.x * 2 + half; r < M; r += (long long)gridDim.x * 2) {
    float v = H[r * D + c];
    s += v; s2 += v * v;
  }
  __shared__ float sh[2 * D];
  if (half) { sh[c] = s; sh[D + c] = s2; }
  __syncthreads();
  if (!half) {
    atomicAdd(&stat[c],     s  + sh[c]);
    atomicAdd(&stat[D + c], s2 + sh[D + c]);
  }
}
__global__ void bn_finalize_kernel(float* stat, const float* g, const float* b, int M) {
  int c = threadIdx.x;
  float mu  = stat[c] / (float)M;
  float var = stat[D + c] / (float)M - mu * mu;
  float sc  = rsqrtf(var + 1e-5f) * g[c];
  stat[2 * D + c] = sc;
  stat[3 * D + c] = b[c] - mu * sc;
}
__global__ void bn_apply_kernel(float* H, const float* __restrict__ stat, long long total) {
  long long i = (long long)blockIdx.x * blockDim.x + threadIdx.x;
  if (i >= total) return;
  int c = (int)(i & (D - 1));
  float v = H[i] * stat[2 * D + c] + stat[3 * D + c];
  H[i] = v > 0.f ? v : 0.f;
}

// GCN message: 32 threads/edge, 4 channels/thread, atomic scatter-add
__global__ void msg_gcn_kernel(const float* __restrict__ xlin,
                               const int* __restrict__ dst, const int* __restrict__ src,
                               const float* __restrict__ dinv, float* msggcn, int E) {
  long long t = (long long)blockIdx.x * blockDim.x + threadIdx.x;
  long long e = t >> 5;
  if (e >= E) return;
  int c = (int)(t & 31) * 4;
  int d = dst[e], s = src[e];
  float w = dinv[d] * dinv[s];
  const float* xr = xlin + (size_t)s * D + c;
  float* mr = msggcn + (size_t)d * D + c;
#pragma unroll
  for (int j = 0; j < 4; ++j) atomicAdd(mr + j, xr[j] * w);
}

// segment softmax pass 1: per-dst per-channel max (encoded-uint atomic max)
__global__ void seg_max_kernel(const float* __restrict__ res, const int* __restrict__ perm,
                               const int* __restrict__ dst, const int* __restrict__ offp,
                               int Rn, unsigned* segmax) {
  long long t = (long long)blockIdx.x * blockDim.x + threadIdx.x;
  long long p = t >> 5;
  if (p >= offp[Rn]) return;
  int e = perm[p]; if (e < 0) return;
  int c = (int)(t & 31) * 4;
  int d = dst[e];
  const float* rr = res + p * D + c;
  unsigned* mm = segmax + (size_t)d * D + c;
#pragma unroll
  for (int j = 0; j < 4; ++j) atomicMax(mm + j, fenc(rr[j]));
}

// pass 2: denom += exp(res - max)
__global__ void seg_expsum_kernel(const float* __restrict__ res, const int* __restrict__ perm,
                                  const int* __restrict__ dst, const int* __restrict__ offp,
                                  int Rn, const unsigned* __restrict__ segmax, float* denom) {
  long long t = (long long)blockIdx.x * blockDim.x + threadIdx.x;
  long long p = t >> 5;
  if (p >= offp[Rn]) return;
  int e = perm[p]; if (e < 0) return;
  int c = (int)(t & 31) * 4;
  int d = dst[e];
  const float* rr = res + p * D + c;
  const unsigned* mm = segmax + (size_t)d * D + c;
  float* dn = denom + (size_t)d * D + c;
#pragma unroll
  for (int j = 0; j < 4; ++j) atomicAdd(dn + j, expf(rr[j] - fdec(mm[j])));
}

// pass 3: msg += res * exp(res - max)/denom  (recompute exp: no 2nd E*D buffer)
__global__ void seg_weighted_kernel(const float* __restrict__ res, const int* __restrict__ perm,
                                    const int* __restrict__ dst, const int* __restrict__ offp,
                                    int Rn, const unsigned* __restrict__ segmax,
                                    const float* __restrict__ denom, float* msg) {
  long long t = (long long)blockIdx.x * blockDim.x + threadIdx.x;
  long long p = t >> 5;
  if (p >= offp[Rn]) return;
  int e = perm[p]; if (e < 0) return;
  int c = (int)(t & 31) * 4;
  int d = dst[e];
  const float* rr = res + p * D + c;
  const unsigned* mm = segmax + (size_t)d * D + c;
  const float* dn = denom + (size_t)d * D + c;
  float* mg = msg + (size_t)d * D + c;
#pragma unroll
  for (int j = 0; j < 4; ++j) {
    float v = rr[j];
    float w = expf(v - fdec(mm[j])) / dn[j];
    atomicAdd(mg + j, v * w);
  }
}

__global__ void combine_kernel(const float* __restrict__ msggcn, const float* __restrict__ msg,
                               float* tmp, long long total) {
  long long i = (long long)blockIdx.x * blockDim.x + threadIdx.x;
  if (i < total) tmp[i] = msggcn[i] + 0.1f * fmaxf(msg[i], 0.f);
}

// exact GELU + row gather
__global__ void gelu_gather_kernel(const float* __restrict__ H, const int* __restrict__ idx,
                                   float* out, int NI) {
  long long t = (long long)blockIdx.x * blockDim.x + threadIdx.x;
  if (t >= (long long)NI * D) return;
  int i = (int)(t >> 7), c = (int)(t & (D - 1));
  float x = H[(size_t)idx[i] * D + c];
  out[t] = 0.5f * x * (1.f + erff(x * 0.70710678118654752f));
}

// ---------------------------------------------------------------------------
extern "C" void kernel_launch(void* const* d_in, const int* in_sizes, int n_in,
                              void* d_out, int out_size, void* d_ws, size_t ws_size,
                              hipStream_t stream) {
  const float* x      = (const float*)d_in[0];
  const int*   ei     = (const int*)d_in[1];
  const int*   idx    = (const int*)d_in[2];
  const int*   etype  = (const int*)d_in[3];
  const float* proj_w = (const float*)d_in[5];
  const float* proj_b = (const float*)d_in[6];
  const float* bn_g   = (const float*)d_in[7];
  const float* bn_b   = (const float*)d_in[8];
  const float* win_w  = (const float*)d_in[9];
  const float* win_b  = (const float*)d_in[10];
  const float* wrel   = (const float*)d_in[11];
  const float* out_w  = (const float*)d_in[12];
  const float* out_b  = (const float*)d_in[13];
  float* out = (float*)d_out;

  const int N  = in_sizes[0] / D;
  const int E  = in_sizes[1] / 2;
  const int NI = in_sizes[2];
  const int L  = in_sizes[10] / D;
  const int Rn = in_sizes[11] / (L * D * D);
  const int EP = E + 16 * Rn;              // padded sorted-edge capacity
  const int WSZ = D * D;                   // packed weight image size (elems)

  const int* dstp = ei;                    // edge_index[0]
  const int* srcp = ei + E;                // edge_index[1]

  // workspace carve-out (256B aligned)
  char* base = (char*)d_ws;
  size_t off = 0;
  auto alloc = [&](size_t bytes) -> void* {
    off = (off + 255) & ~(size_t)255;
    void* p = base + off; off += bytes; return p;
  };
  const size_t ND = (size_t)N * D;
  float*    hA     = (float*)alloc(ND * 4);
  float*    hB     = (float*)alloc(ND * 4);
  float*    xlin   = (float*)alloc(ND * 4);
  float*    tmpin  = (float*)alloc(ND * 4);
  float*    msg    = (float*)alloc(ND * 4);
  float*    msggcn = (float*)alloc(ND * 4);
  float*    denom  = (float*)alloc(ND * 4);
  unsigned* segmax = (unsigned*)alloc(ND * 4);
  __bf16*   pbufA  = (__bf16*)alloc(ND * 2);            // packed A rows (reused)
  __bf16*   pw_all = (__bf16*)alloc((size_t)(1 + 2 * L + L * Rn) * WSZ * 2);
  float*    deg    = (float*)alloc((size_t)N * 4);
  float*    dinv   = (float*)alloc((size_t)N * 4);
  float*    stat   = (float*)alloc(4 * D * 4);
  int*      cnt    = (int*)alloc(Rn * 4);
  int*      cursor = (int*)alloc(Rn * 4);
  int*      offp   = (int*)alloc((Rn + 1) * 4);
  int*      perm   = (int*)alloc((size_t)EP * 4);
  float*    res    = (float*)alloc((size_t)EP * D * 4);
  (void)ws_size; (void)n_in; (void)out_size;

  // packed weight images: [proj][win 0..L-1][out 0..L-1][rel 0..L*R-1]
  __bf16* pw_proj = pw_all;
  __bf16* pw_win  = pw_proj + WSZ;
  __bf16* pw_out  = pw_win + (size_t)L * WSZ;
  __bf16* pw_rel  = pw_out + (size_t)L * WSZ;

  auto blocks = [](long long n, int t) { return (unsigned)((n + t - 1) / t); };
  const int mtiles = (N + 15) / 16;
  const unsigned wb = blocks(WSZ, 256);

  // ---- pack all weight matrices to bf16 fragment order ----
  pack_w_kernel<<<wb, 256, 0, stream>>>(proj_w, pw_proj);
  for (int l = 0; l < L; ++l) {
    pack_w_kernel<<<wb, 256, 0, stream>>>(win_w + (size_t)l * WSZ, pw_win + (size_t)l * WSZ);
    pack_w_kernel<<<wb, 256, 0, stream>>>(out_w + (size_t)l * WSZ, pw_out + (size_t)l * WSZ);
    for (int r = 0; r < Rn; ++r)
      pack_w_kernel<<<wb, 256, 0, stream>>>(wrel + ((size_t)l * Rn + r) * WSZ,
                                            pw_rel + ((size_t)l * Rn + r) * WSZ);
  }

  // ---- edge bucketing by relation (16-aligned padded) ----
  fill_u32_kernel<<<blocks(Rn, 256), 256, 0, stream>>>((unsigned*)cnt, Rn, 0u);
  fill_u32_kernel<<<blocks(Rn, 256), 256, 0, stream>>>((unsigned*)cursor, Rn, 0u);
  fill_u32_kernel<<<blocks(EP, 256), 256, 0, stream>>>((unsigned*)perm, EP, 0xFFFFFFFFu);
  count_kernel<<<blocks(E, 256), 256, 0, stream>>>(etype, cnt, E);
  offsets_kernel<<<1, 1, 0, stream>>>(cnt, offp, Rn);
  scatter_kernel<<<blocks(E, 256), 256, 0, stream>>>(etype, cursor, offp, perm, E);

  // ---- degrees / gcn norm ----
  fill_f32_kernel<<<blocks(N, 256), 256, 0, stream>>>(deg, N, 0.f);
  deg_kernel<<<blocks(E, 256), 256, 0, stream>>>(srcp, deg, E);
  dinv_kernel<<<blocks(N, 256), 256, 0, stream>>>(deg, dinv, N);

  // ---- projection + BN + ReLU ----
  pack_a_kernel<<<blocks((long long)ND, 256), 256, 0, stream>>>(x, pbufA, (long long)ND);
  gemm_bias_kernel<<<mtiles, 256, 0, stream>>>(pbufA, pw_proj, proj_b, hA, N);
  fill_f32_kernel<<<blocks(4 * D, 256), 256, 0, stream>>>(stat, 4 * D, 0.f);
  bn_stats_kernel<<<512, 256, 0, stream>>>(hA, stat, N);
  bn_finalize_kernel<<<1, D, 0, stream>>>(stat, bn_g, bn_b, N);
  bn_apply_kernel<<<blocks((long long)ND, 256), 256, 0, stream>>>(hA, stat, (long long)ND);

  // ---- DAN layers ----
  float* hin = hA;
  float* hout = hB;
  for (int l = 0; l < L; ++l) {
    pack_a_kernel<<<blocks((long long)ND, 256), 256, 0, stream>>>(hin, pbufA, (long long)ND);
    gemm_bias_kernel<<<mtiles, 256, 0, stream>>>(pbufA, pw_win + (size_t)l * WSZ,
                                                 win_b + (size_t)l * D, xlin, N);
    fill_f32_kernel<<<blocks((long long)ND, 256), 256, 0, stream>>>(msggcn, (long long)ND, 0.f);
    fill_f32_kernel<<<blocks((long long)ND, 256), 256, 0, stream>>>(msg,    (long long)ND, 0.f);
    fill_f32_kernel<<<blocks((long long)ND, 256), 256, 0, stream>>>(denom,  (long long)ND, 0.f);
    fill_u32_kernel<<<blocks((long long)ND, 256), 256, 0, stream>>>(segmax, (long long)ND, 0u);

    msg_gcn_kernel<<<blocks((long long)E * 32, 256), 256, 0, stream>>>(
        xlin, dstp, srcp, dinv, msggcn, E);

    pack_a_kernel<<<blocks((long long)ND, 256), 256, 0, stream>>>(xlin, pbufA, (long long)ND);
    rel_gemm_kernel<<<EP / 16, 256, 0, stream>>>(
        pbufA, pw_rel + (size_t)l * Rn * WSZ, srcp, perm, offp, cnt, Rn, res);

    seg_max_kernel<<<blocks((long long)EP * 32, 256), 256, 0, stream>>>(
        res, perm, dstp, offp, Rn, segmax);
    seg_expsum_kernel<<<blocks((long long)EP * 32, 256), 256, 0, stream>>>(
        res, perm, dstp, offp, Rn, segmax, denom);
    seg_weighted_kernel<<<blocks((long long)EP * 32, 256), 256, 0, stream>>>(
        res, perm, dstp, offp, Rn, segmax, denom, msg);

    combine_kernel<<<blocks((long long)ND, 256), 256, 0, stream>>>(
        msggcn, msg, tmpin, (long long)ND);
    pack_a_kernel<<<blocks((long long)ND, 256), 256, 0, stream>>>(tmpin, pbufA, (long long)ND);
    gemm_bias_kernel<<<mtiles, 256, 0, stream>>>(pbufA, pw_out + (size_t)l * WSZ,
                                                 out_b + (size_t)l * D, hout, N);
    float* t = hin; hin = hout; hout = t;
  }

  // ---- GELU + gather ----
  gelu_gather_kernel<<<blocks((long long)NI * D, 256), 256, 0, stream>>>(hin, idx, out, NI);
}
